// GPT2_TransformerBlock_472446403097
// MI455X (gfx1250) — matrix operand order repaired and checked by reading, not compiled, not run on hardware
//
#include <hip/hip_runtime.h>
#include <math.h>
#include <stdint.h>

typedef __bf16 bf16;
typedef __attribute__((ext_vector_type(16))) __bf16 v16bf;
typedef __attribute__((ext_vector_type(8)))  float  v8f;
typedef int v4i_ __attribute__((vector_size(16)));

#define D_MODEL 768
#define N_HEADQ 12
#define DK      64
#define D_FF    3072
#define SEQ     2048
#define NBATCH  4
#define ROWS    (NBATCH * SEQ)   // 8192

// ---- CDNA5 async copy to LDS (guarded: fall back to sync staging if absent) ----
#if defined(__has_builtin)
#if __has_builtin(__builtin_amdgcn_global_load_async_to_lds_b128) && \
    __has_builtin(__builtin_amdgcn_s_wait_asynccnt)
#define USE_ASYNC_LDS 1
#endif
#endif
#ifndef USE_ASYNC_LDS
#define USE_ASYNC_LDS 0
#endif

#define AS1P(p) ((__attribute__((address_space(1))) v4i_*)(uintptr_t)(p))
#define AS3P(p) ((__attribute__((address_space(3))) v4i_*)(unsigned)(uintptr_t)(p))

// ---------------------------------------------------------------------------
// Generic batched bf16 WMMA GEMM:  C[z] = epilogue( A[z] (MxK) * Bt[z]^T )
//   A: row-major bf16 (AF32=1: f32, converted while staging to LDS)
//   Bt: row-major [N x K] bf16 (B transposed) -> matches B fragment gather
//   block tile 128(M) x 64(N), 8 waves, each wave owns 16 rows x 64 cols
// ---------------------------------------------------------------------------
template <int AF32>
__global__ __launch_bounds__(256)
void gemm_bf16_wmma(const void* __restrict__ Av, int lda,
                    long long a_sb, long long a_sh,
                    const bf16* __restrict__ Bt, int ldb,
                    long long b_sb, long long b_sh,
                    const float* __restrict__ bias,
                    float* __restrict__ Cf, bf16* __restrict__ Cb, int ldc,
                    long long c_sb, long long c_sh,
                    const float* __restrict__ resid,
                    int Hdiv, int K, float scale, int causal, int do_gelu)
{
#if USE_ASYNC_LDS
    constexpr int NBUF = AF32 ? 1 : 2;     // double-buffer the async path
#else
    constexpr int NBUF = 1;
#endif
    __shared__ bf16 As[NBUF][128 * 32];
    __shared__ bf16 Bs[NBUF][64 * 32];

    const int tid  = threadIdx.x;
    const int wave = tid >> 5;
    const int lane = tid & 31;
    const int half = lane >> 4;   // which 16-lane half of the wave
    const int l16  = lane & 15;

    const int z  = blockIdx.z;
    const int zb = z / Hdiv;
    const int zh = z % Hdiv;
    const long long aoff = (long long)zb * a_sb + (long long)zh * a_sh;
    const long long boff = (long long)zb * b_sb + (long long)zh * b_sh;
    const long long coff = (long long)zb * c_sb + (long long)zh * c_sh;

    const int row0 = blockIdx.y * 128;
    const int col0 = blockIdx.x * 64;

    // per-thread staging coordinates (constant across K chunks)
    const int ar = (tid * 16) >> 5, ac = (tid * 16) & 31;   // A: 16 elems (half row)
    const int br = (tid * 8) >> 5,  bc = (tid * 8) & 31;    // B: 8 elems

    const float* Af = (const float*)Av;
    const bf16*  Ab = (const bf16*)Av;

    v8f acc[4];
#pragma unroll
    for (int t = 0; t < 4; ++t)
#pragma unroll
        for (int e = 0; e < 8; ++e) acc[t][e] = 0.0f;

    auto compute = [&](int p) {
        // A fragment: 16x32 bf16, ISA layout (K 0..7 & 16..23 | 8..15 & 24..31)
        union { uint4 u[2]; v16bf v; } fa;
        const uint4* A4 = (const uint4*)&As[p][(wave * 16 + l16) * 32];
        fa.u[0] = A4[half];
        fa.u[1] = A4[2 + half];
#pragma unroll
        for (int t = 0; t < 4; ++t) {
            // B fragment: 32x16, ISA layout (K 0..7 & 16..23 | 8..15 & 24..31)
            union { uint4 u[2]; v16bf v; } fb;
            const uint4* B4 = (const uint4*)&Bs[p][(t * 16 + l16) * 32];
            fb.u[0] = B4[half];
            fb.u[1] = B4[2 + half];
            acc[t] = __builtin_amdgcn_wmma_f32_16x16x32_bf16(
                false, fa.v, false, fb.v, (short)0, acc[t], false, false);
        }
    };

    if constexpr (!AF32) {
#if USE_ASYNC_LDS
        // ---- async double-buffered pipeline: 3 async b128 per thread/stage ----
        auto issue = [&](int p, int k0) {
            const bf16* ga = Ab + aoff + (long long)(row0 + ar) * lda + k0 + ac;
            bf16* la = &As[p][ar * 32 + ac];
            __builtin_amdgcn_global_load_async_to_lds_b128(AS1P(ga), AS3P(la), 0, 0);
            __builtin_amdgcn_global_load_async_to_lds_b128(AS1P(ga), AS3P(la), 16, 0);
            const bf16* gb = Bt + boff + (long long)(col0 + br) * ldb + k0 + bc;
            bf16* lb = &Bs[p][br * 32 + bc];
            __builtin_amdgcn_global_load_async_to_lds_b128(AS1P(gb), AS3P(lb), 0, 0);
        };
        const int nch = K >> 5;
        issue(0, 0);
        for (int i = 0; i < nch; ++i) {
            const int p = i & 1;
            if (i + 1 < nch) {
                issue(p ^ 1, (i + 1) << 5);
                __builtin_amdgcn_s_wait_asynccnt(3);  // stage i done (in-order)
            } else {
                __builtin_amdgcn_s_wait_asynccnt(0);
            }
            __syncthreads();
            compute(p);
            __syncthreads();
        }
#else
        for (int k0 = 0; k0 < K; k0 += 32) {
            const uint4* s4 =
                (const uint4*)(Ab + aoff + (long long)(row0 + ar) * lda + k0 + ac);
            uint4* d4 = (uint4*)&As[0][ar * 32 + ac];
            d4[0] = s4[0];
            d4[1] = s4[1];
            const uint4* sb =
                (const uint4*)(Bt + boff + (long long)(col0 + br) * ldb + k0 + bc);
            *(uint4*)&Bs[0][br * 32 + bc] = sb[0];
            __syncthreads();
            compute(0);
            __syncthreads();
        }
#endif
    } else {
        // ---- f32 A (softmax probabilities): convert to bf16 while staging ----
        for (int k0 = 0; k0 < K; k0 += 32) {
            const float4* s4 =
                (const float4*)(Af + aoff + (long long)(row0 + ar) * lda + k0 + ac);
            bf16* dst = &As[0][ar * 32 + ac];
#pragma unroll
            for (int q = 0; q < 4; ++q) {
                float4 f = s4[q];
                dst[q * 4 + 0] = (bf16)f.x;
                dst[q * 4 + 1] = (bf16)f.y;
                dst[q * 4 + 2] = (bf16)f.z;
                dst[q * 4 + 3] = (bf16)f.w;
            }
            const uint4* sb =
                (const uint4*)(Bt + boff + (long long)(col0 + br) * ldb + k0 + bc);
            *(uint4*)&Bs[0][br * 32 + bc] = sb[0];
            __syncthreads();
            compute(0);
            __syncthreads();
        }
    }

    // ---- epilogue: C/D layout -> row = r + 8*half (per wave 16-row strip) ----
    const int rbase = row0 + wave * 16 + half * 8;
#pragma unroll
    for (int t = 0; t < 4; ++t) {
        const int col = col0 + t * 16 + l16;
        const float bv = bias ? bias[col] : 0.0f;
#pragma unroll
        for (int r = 0; r < 8; ++r) {
            const int row = rbase + r;
            float v = acc[t][r] * scale + bv;
            if (causal && col > row) v += -10000.0f;
            if (do_gelu) {
                float u = v;
                v = 0.5f * u *
                    (1.0f + tanhf(0.7978845608028654f * (u + 0.044715f * u * u * u)));
            }
            if (resid) v += resid[(long long)row * ldc + col];
            const long long ci = coff + (long long)row * ldc + col;
            if (Cf) Cf[ci] = v;
            if (Cb) Cb[ci] = (bf16)v;
        }
    }
}

// ---------------------------------------------------------------------------
// LayerNorm over 768-wide rows -> bf16
// ---------------------------------------------------------------------------
__global__ __launch_bounds__(256)
void layernorm_to_bf16(const float* __restrict__ x, const float* __restrict__ g,
                       const float* __restrict__ b, bf16* __restrict__ out)
{
    __shared__ float red[256];
    const int tid = threadIdx.x;
    const long long row = blockIdx.x;
    const float* xr = x + row * D_MODEL;

    float local[3];
    float s = 0.0f;
#pragma unroll
    for (int i = 0; i < 3; ++i) { local[i] = xr[tid + i * 256]; s += local[i]; }
    red[tid] = s;
    __syncthreads();
    for (int off = 128; off > 0; off >>= 1) {
        if (tid < off) red[tid] += red[tid + off];
        __syncthreads();
    }
    const float mu = red[0] * (1.0f / D_MODEL);
    __syncthreads();

    float vs = 0.0f;
#pragma unroll
    for (int i = 0; i < 3; ++i) { float d = local[i] - mu; vs += d * d; }
    red[tid] = vs;
    __syncthreads();
    for (int off = 128; off > 0; off >>= 1) {
        if (tid < off) red[tid] += red[tid + off];
        __syncthreads();
    }
    const float rstd = rsqrtf(red[0] * (1.0f / D_MODEL) + 1e-5f);

#pragma unroll
    for (int i = 0; i < 3; ++i) {
        const int c = tid + i * 256;
        out[row * D_MODEL + c] = (bf16)((local[i] - mu) * rstd * g[c] + b[c]);
    }
}

// ---------------------------------------------------------------------------
// Row softmax over 2048-wide rows, in place (f32)
// ---------------------------------------------------------------------------
__global__ __launch_bounds__(256)
void softmax_rows(float* __restrict__ attn)
{
    __shared__ float red[256];
    const int tid = threadIdx.x;
    float* p = attn + (long long)blockIdx.x * SEQ;

    float v[8];
    float m = -1e30f;
#pragma unroll
    for (int j = 0; j < 8; ++j) { v[j] = p[tid + j * 256]; m = fmaxf(m, v[j]); }
    red[tid] = m;
    __syncthreads();
    for (int off = 128; off > 0; off >>= 1) {
        if (tid < off) red[tid] = fmaxf(red[tid], red[tid + off]);
        __syncthreads();
    }
    const float rowmax = red[0];
    __syncthreads();

    float s = 0.0f;
#pragma unroll
    for (int j = 0; j < 8; ++j) { v[j] = __expf(v[j] - rowmax); s += v[j]; }
    red[tid] = s;
    __syncthreads();
    for (int off = 128; off > 0; off >>= 1) {
        if (tid < off) red[tid] += red[tid + off];
        __syncthreads();
    }
    const float inv = 1.0f / red[0];
#pragma unroll
    for (int j = 0; j < 8; ++j) p[tid + j * 256] = v[j] * inv;
}

// ---------------------------------------------------------------------------
// Split fused QKV (bf16 [ROWS,2304]) -> q[b,h,s,d], k[b,h,s,d], vT[b,h,d,s]
// ---------------------------------------------------------------------------
__global__ __launch_bounds__(256)
void split_qkv(const bf16* __restrict__ qkv, bf16* __restrict__ q,
               bf16* __restrict__ k, bf16* __restrict__ vT)
{
    const long long i = (long long)blockIdx.x * 256 + threadIdx.x;
    if (i >= (long long)ROWS * 2304) return;
    const int col = (int)(i % 2304);
    const long long row = i / 2304;
    const int b = (int)(row / SEQ);
    const int s = (int)(row % SEQ);
    const int which = col / D_MODEL;
    const int h = (col % D_MODEL) / DK;
    const int d = col % DK;
    const bf16 val = qkv[i];
    const long long bh = (long long)(b * N_HEADQ + h);
    if (which == 0)      q[(bh * SEQ + s) * DK + d] = val;
    else if (which == 1) k[(bh * SEQ + s) * DK + d] = val;
    else                 vT[(bh * DK + d) * SEQ + s] = val;
}

// ---------------------------------------------------------------------------
// W[K,N] f32 -> Wt[N,K] bf16 (so B is row-major in N for the GEMM)
// ---------------------------------------------------------------------------
__global__ __launch_bounds__(256)
void transpose_to_bf16(const float* __restrict__ W, bf16* __restrict__ Wt,
                       int Kd, int Nd)
{
    const long long i = (long long)blockIdx.x * 256 + threadIdx.x;
    if (i >= (long long)Kd * Nd) return;
    const int n = (int)(i % Nd);
    const long long kk = i / Nd;
    Wt[(long long)n * Kd + kk] = (bf16)W[i];
}

// ---------------------------------------------------------------------------
extern "C" void kernel_launch(void* const* d_in, const int* in_sizes, int n_in,
                              void* d_out, int out_size, void* d_ws, size_t ws_size,
                              hipStream_t stream)
{
    (void)in_sizes; (void)n_in; (void)out_size; (void)ws_size;

    const float* x    = (const float*)d_in[0];
    // d_in[1] = look_ahead_mask (causal; handled analytically in the epilogue)
    const float* Wqkv = (const float*)d_in[2];
    const float* bqkv = (const float*)d_in[3];
    const float* Wo   = (const float*)d_in[4];
    const float* bo   = (const float*)d_in[5];
    const float* Wfc  = (const float*)d_in[6];
    const float* bfc  = (const float*)d_in[7];
    const float* Wpr  = (const float*)d_in[8];
    const float* bpr  = (const float*)d_in[9];
    const float* g1   = (const float*)d_in[10];
    const float* b1   = (const float*)d_in[11];
    const float* g2   = (const float*)d_in[12];
    const float* b2   = (const float*)d_in[13];

    float* xout = (float*)d_out;                                  // [4,2048,768]
    float* attn = (float*)d_out + (long long)ROWS * D_MODEL;      // [4,12,2048,2048]

    // ---- workspace carve-up (256B aligned) ----
    char*  wsb = (char*)d_ws;
    size_t off = 0;
    auto carve = [&](size_t bytes) -> char* {
        char* p = wsb + off;
        off = (off + bytes + 255) & ~(size_t)255;
        return p;
    };
    bf16*  wqkv_t  = (bf16*)carve((size_t)2304 * 768 * 2);
    bf16*  wo_t    = (bf16*)carve((size_t)768 * 768 * 2);
    bf16*  wfc_t   = (bf16*)carve((size_t)3072 * 768 * 2);
    bf16*  wpr_t   = (bf16*)carve((size_t)768 * 3072 * 2);
    bf16*  nx      = (bf16*)carve((size_t)ROWS * 768 * 2);
    bf16*  qkv     = (bf16*)carve((size_t)ROWS * 2304 * 2);
    bf16*  qb      = (bf16*)carve((size_t)ROWS * 768 * 2);
    bf16*  kb      = (bf16*)carve((size_t)ROWS * 768 * 2);
    bf16*  vT      = (bf16*)carve((size_t)ROWS * 768 * 2);
    bf16*  attnout = (bf16*)carve((size_t)ROWS * 768 * 2);
    float* x1      = (float*)carve((size_t)ROWS * 768 * 4);
    bf16*  h2      = (bf16*)carve((size_t)ROWS * 768 * 2);
    bf16*  m2      = (bf16*)carve((size_t)ROWS * 3072 * 2);

    const dim3 blk(256);

    // 1. weight transposes -> bf16
    transpose_to_bf16<<<dim3((768u * 2304u + 255u) / 256u), blk, 0, stream>>>(Wqkv, wqkv_t, 768, 2304);
    transpose_to_bf16<<<dim3((768u * 768u  + 255u) / 256u), blk, 0, stream>>>(Wo,   wo_t,   768, 768);
    transpose_to_bf16<<<dim3((768u * 3072u + 255u) / 256u), blk, 0, stream>>>(Wfc,  wfc_t,  768, 3072);
    transpose_to_bf16<<<dim3((3072u * 768u + 255u) / 256u), blk, 0, stream>>>(Wpr,  wpr_t,  3072, 768);

    // 2. LN1: x -> nx (bf16)
    layernorm_to_bf16<<<dim3(ROWS), blk, 0, stream>>>(x, g1, b1, nx);

    // 3. QKV GEMM: [8192,768] x [768,2304] + bqkv -> qkv (bf16)
    gemm_bf16_wmma<0><<<dim3(2304 / 64, ROWS / 128, 1), blk, 0, stream>>>(
        nx, 768, 0, 0, wqkv_t, 768, 0, 0, bqkv,
        nullptr, qkv, 2304, 0, 0, nullptr, 1, 768, 1.0f, 0, 0);

    // 4. split into q,k,vT (per-head layouts)
    split_qkv<<<dim3((unsigned)(((long long)ROWS * 2304 + 255) / 256)), blk, 0, stream>>>(
        qkv, qb, kb, vT);

    // 5. scores = q @ k^T * 1/8 + causal  -> attn (f32, in d_out)
    gemm_bf16_wmma<0><<<dim3(SEQ / 64, SEQ / 128, NBATCH * N_HEADQ), blk, 0, stream>>>(
        qb, DK, (long long)SEQ * DK, 0,
        kb, DK, (long long)SEQ * DK, 0, nullptr,
        attn, nullptr, SEQ, (long long)SEQ * SEQ, 0,
        nullptr, 1, DK, 0.125f, 1, 0);

    // 6. softmax rows, in place
    softmax_rows<<<dim3(NBATCH * N_HEADQ * SEQ), blk, 0, stream>>>(attn);

    // 7. out = P @ V : A is f32 attn (converted in-flight), scatter heads to [B,S,D]
    gemm_bf16_wmma<1><<<dim3(DK / 64, SEQ / 128, NBATCH * N_HEADQ), blk, 0, stream>>>(
        attn, SEQ, (long long)N_HEADQ * SEQ * SEQ, (long long)SEQ * SEQ,
        vT, SEQ, (long long)N_HEADQ * DK * SEQ, (long long)DK * SEQ, nullptr,
        nullptr, attnout, D_MODEL, (long long)SEQ * D_MODEL, DK,
        nullptr, N_HEADQ, SEQ, 1.0f, 0, 0);

    // 8. a = out @ Wo + bo ; x1 = x + a  (f32)
    gemm_bf16_wmma<0><<<dim3(768 / 64, ROWS / 128, 1), blk, 0, stream>>>(
        attnout, 768, 0, 0, wo_t, 768, 0, 0, bo,
        x1, nullptr, 768, 0, 0, x, 1, 768, 1.0f, 0, 0);

    // 9. LN2: x1 -> h2 (bf16)
    layernorm_to_bf16<<<dim3(ROWS), blk, 0, stream>>>(x1, g2, b2, h2);

    // 10. m = gelu(h2 @ Wfc + bfc) (bf16)
    gemm_bf16_wmma<0><<<dim3(3072 / 64, ROWS / 128, 1), blk, 0, stream>>>(
        h2, 768, 0, 0, wfc_t, 768, 0, 0, bfc,
        nullptr, m2, 3072, 0, 0, nullptr, 1, 768, 1.0f, 0, 1);

    // 11. xout = x1 + m @ Wpr + bpr (f32, final output)
    gemm_bf16_wmma<0><<<dim3(768 / 64, ROWS / 128, 1), blk, 0, stream>>>(
        m2, 3072, 0, 0, wpr_t, 3072, 0, 0, bpr,
        xout, nullptr, 768, 0, 0, x1, 1, 3072, 1.0f, 0, 0);
}
